// AdaptiveGCN_7670811591018
// MI455X (gfx1250) — matrix operand order repaired and checked by reading, not compiled, not run on hardware
//
#include <hip/hip_runtime.h>
#include <math.h>

#define D 128
#define BN_EPS 1e-5f

typedef __attribute__((ext_vector_type(16))) _Float16 v16h;
typedef __attribute__((ext_vector_type(8)))  float    v8f;

// Native L2 FP32 atomic add (non-returning form) — avoids any CAS-loop expansion.
__device__ __forceinline__ void atomic_add_f32(float* p, float v) {
    asm volatile("global_atomic_add_f32 %0, %1, off" :: "v"(p), "v"(v) : "memory");
}

// ---------------- degree / norm ----------------
__global__ void k_init_deg(float* deg, int n) {
    int i = blockIdx.x * blockDim.x + threadIdx.x;
    if (i < n) deg[i] = 1.0f;                       // self-loop weight 1
}

__global__ void k_accum_deg(const int* __restrict__ dst, const float* __restrict__ ew,
                            float* deg, int e) {
    int i = blockIdx.x * blockDim.x + threadIdx.x;
    if (i < e) atomic_add_f32(&deg[dst[i]], ew[i]);
}

__global__ void k_make_dinv(float* deg, int n) {
    int i = blockIdx.x * blockDim.x + threadIdx.x;
    if (i < n) deg[i] = rsqrtf(deg[i]);             // deg >= 1 always (self-loop)
}

// ---------------- GEMM: H = X(f32,[n,128]) @ W(f32,[128,128]) via WMMA f16 ----------------
// 256 threads = 8 waves; wave w computes the 16x16 tile (rowblock, coltile=w).
// Epilogue also writes AGG = bias + H*dinv^2 (self-loop term of the aggregation).
// W staged in LDS pre-swizzled to the CDNA5 B-operand layout:
//   lane l (kg=l>>4, nlane=l&15) holds 16 halves = W[kstep*32+kg*8+(j&7)+(j>>3)*16][ntile*16+nlane]
__global__ __launch_bounds__(256) void k_gemm_wmma(const float* __restrict__ X,
                                                   const float* __restrict__ W,
                                                   const float* __restrict__ dinv,
                                                   const float* __restrict__ bias,
                                                   float* __restrict__ H,
                                                   float* __restrict__ AGG, int n) {
    __shared__ __align__(32) _Float16 ldsB[4 * 8 * 32 * 16];   // 32 KB

    const int tid = threadIdx.x;
    // ---- stage W into LDS (each thread fills 4 lane-vectors of 16 halves) ----
    #pragma unroll
    for (int q = 0; q < 4; ++q) {
        int v     = tid * 4 + q;          // 0..1023
        int kstep = v >> 8;
        int rem   = v & 255;
        int ntile = rem >> 5;
        int l     = rem & 31;
        int kg    = l >> 4;
        int nn    = ntile * 16 + (l & 15);
        int kbase = kstep * 32 + kg * 8;
        _Float16* dp = &ldsB[v * 16];
        #pragma unroll
        for (int j = 0; j < 16; ++j)
            dp[j] = (_Float16)W[(kbase + j + (j >> 3) * 8) * D + nn];
    }
    __syncthreads();

    const int wave = tid >> 5;            // column tile 0..7
    const int lane = tid & 31;
    const int kg   = lane >> 4;
    const int mr   = lane & 15;
    int row = blockIdx.x * 16 + mr;
    if (row >= n) row = n - 1;            // keep EXEC all-ones for WMMA; tail stores guarded below

    v8f c = {};
    #pragma unroll
    for (int kstep = 0; kstep < 4; ++kstep) {
        // A operand: lanes 0-15 row m, K = {kg*8..+7} and {kg*8+16..+23} within this 32-chunk
        const float* xr = X + (size_t)row * D + kstep * 32 + kg * 8;
        float4 p0 = *(const float4*)(xr);
        float4 p1 = *(const float4*)(xr + 4);
        float4 p2 = *(const float4*)(xr + 16);
        float4 p3 = *(const float4*)(xr + 20);
        v16h a;
        a[0]=(_Float16)p0.x; a[1]=(_Float16)p0.y; a[2]=(_Float16)p0.z; a[3]=(_Float16)p0.w;
        a[4]=(_Float16)p1.x; a[5]=(_Float16)p1.y; a[6]=(_Float16)p1.z; a[7]=(_Float16)p1.w;
        a[8]=(_Float16)p2.x; a[9]=(_Float16)p2.y; a[10]=(_Float16)p2.z; a[11]=(_Float16)p2.w;
        a[12]=(_Float16)p3.x; a[13]=(_Float16)p3.y; a[14]=(_Float16)p3.z; a[15]=(_Float16)p3.w;

        v16h b = *(const v16h*)&ldsB[(((size_t)kstep * 8 + wave) * 32 + lane) * 16];

        c = __builtin_amdgcn_wmma_f32_16x16x32_f16(false, a, false, b, (short)0, c, false, false);
    }

    // C/D layout: VGPR r -> row m0 + kg*8 + r, col = wave*16 + (lane&15)
    const int col  = wave * 16 + mr;
    const float bc = bias[col];
    #pragma unroll
    for (int r = 0; r < 8; ++r) {
        int m = blockIdx.x * 16 + kg * 8 + r;
        if (m < n) {
            float hv = c[r];
            float di = dinv[m];
            H  [(size_t)m * D + col] = hv;
            AGG[(size_t)m * D + col] = bc + hv * di * di;
        }
    }
}

// ---------------- edge scatter: one wave per edge, float4 per lane, native f32 atomics ----------------
__global__ __launch_bounds__(256) void k_scatter_edges(const float* __restrict__ h,
                                                       const int* __restrict__ src,
                                                       const int* __restrict__ dst,
                                                       const float* __restrict__ ew,
                                                       const float* __restrict__ dinv,
                                                       float* __restrict__ out, int e) {
    int wave = (int)((blockIdx.x * blockDim.x + threadIdx.x) >> 5);
    int lane = threadIdx.x & 31;
    if (wave >= e) return;
    int s = src[wave], d = dst[wave];
    float w = dinv[s] * ew[wave] * dinv[d];
    float4 v = ((const float4*)(h + (size_t)s * D))[lane];
    float* op = out + (size_t)d * D + lane * 4;
    float vx = v.x * w, vy = v.y * w, vz = v.z * w, vw = v.w * w;
    asm volatile("global_atomic_add_f32 %0, %1, off\n\t"
                 "global_atomic_add_f32 %0, %2, off offset:4\n\t"
                 "global_atomic_add_f32 %0, %3, off offset:8\n\t"
                 "global_atomic_add_f32 %0, %4, off offset:12"
                 :: "v"(op), "v"(vx), "v"(vy), "v"(vz), "v"(vw) : "memory");
}

// ---------------- BN (eval) + adaptive ReLU/GELU blend ----------------
__global__ void k_bn_act(const float* __restrict__ g,
                         const float* __restrict__ gamma, const float* __restrict__ beta,
                         const float* __restrict__ mean,  const float* __restrict__ var,
                         const float* __restrict__ ap, float* __restrict__ o, int total) {
    int i = blockIdx.x * blockDim.x + threadIdx.x;
    if (i >= total) return;
    int d = i & 127;
    float v = (g[i] - mean[d]) * rsqrtf(var[d] + BN_EPS) * gamma[d] + beta[d];
    float alpha = 1.0f / (1.0f + expf(-ap[0]));
    float relu = fmaxf(v, 0.0f);
    float gelu = 0.5f * v * (1.0f + erff(v * 0.70710678118654752440f));
    o[i] = alpha * relu + (1.0f - alpha) * gelu;
}

// ---------------- final BN (in-place on d_out) ----------------
__global__ void k_bn_final(float* __restrict__ o,
                           const float* __restrict__ gamma, const float* __restrict__ beta,
                           const float* __restrict__ mean,  const float* __restrict__ var,
                           int total) {
    int i = blockIdx.x * blockDim.x + threadIdx.x;
    if (i >= total) return;
    int d = i & 127;
    o[i] = (o[i] - mean[d]) * rsqrtf(var[d] + BN_EPS) * gamma[d] + beta[d];
}

extern "C" void kernel_launch(void* const* d_in, const int* in_sizes, int n_in,
                              void* d_out, int out_size, void* d_ws, size_t ws_size,
                              hipStream_t stream) {
    (void)n_in; (void)out_size; (void)ws_size;
    const float* x      = (const float*)d_in[0];
    const int*   ei     = (const int*)  d_in[1];
    const float* ew     = (const float*)d_in[2];
    const float* W0     = (const float*)d_in[3];
    const float* b0     = (const float*)d_in[4];
    const float* W1     = (const float*)d_in[5];
    const float* b1     = (const float*)d_in[6];
    const float* gamma0 = (const float*)d_in[7];
    const float* beta0  = (const float*)d_in[8];
    const float* mean0  = (const float*)d_in[9];
    const float* var0   = (const float*)d_in[10];
    const float* gamma1 = (const float*)d_in[11];
    const float* beta1  = (const float*)d_in[12];
    const float* mean1  = (const float*)d_in[13];
    const float* var1   = (const float*)d_in[14];
    const float* ap     = (const float*)d_in[15];

    const int n = in_sizes[0] / D;
    const int e = in_sizes[2];
    const int* src = ei;
    const int* dst = ei + e;

    // workspace: dinv[n] | h[n*D] | g[n*D]
    float* dinv = (float*)d_ws;
    float* h    = dinv + (((size_t)n + 255) & ~(size_t)255);
    float* g    = h + (size_t)n * D;
    float* out  = (float*)d_out;

    const int total = n * D;
    dim3 blk(256);
    dim3 gN((n + 255) / 256), gE((e + 255) / 256), gT((total + 255) / 256);
    dim3 gGemm((n + 15) / 16);
    dim3 gScat((e + 7) / 8);   // 8 waves (edges) per 256-thread block

    // degree + normalization (shared by both layers)
    k_init_deg  <<<gN, blk, 0, stream>>>(dinv, n);
    k_accum_deg <<<gE, blk, 0, stream>>>(dst, ew, dinv, e);
    k_make_dinv <<<gN, blk, 0, stream>>>(dinv, n);

    // layer 0: GEMM (+ fused agg init) -> edge scatter -> BN + adaptive act
    k_gemm_wmma    <<<gGemm, blk, 0, stream>>>(x, W0, dinv, b0, h, g, n);
    k_scatter_edges<<<gScat, blk, 0, stream>>>(h, src, dst, ew, dinv, g, e);
    k_bn_act       <<<gT, blk, 0, stream>>>(g, gamma0, beta0, mean0, var0, ap, h, total);

    // layer 1: GEMM (+ fused agg init into d_out) -> edge scatter -> final BN in-place
    k_gemm_wmma    <<<gGemm, blk, 0, stream>>>(h, W1, dinv, b1, g, out, n);
    k_scatter_edges<<<gScat, blk, 0, stream>>>(g, src, dst, ew, dinv, out, e);
    k_bn_final     <<<gT, blk, 0, stream>>>(out, gamma1, beta1, mean1, var1, total);
}
